// MultiHeadFullyConnectedSelfAttention_48215302865279
// MI455X (gfx1250) — matrix operand order repaired and checked by reading, not compiled, not run on hardware
//
#include <hip/hip_runtime.h>
#include <hip/hip_bf16.h>
#include <math.h>

// ---------------------------------------------------------------------------
// Types for CDNA5 WMMA (wave32)
// ---------------------------------------------------------------------------
typedef __attribute__((ext_vector_type(16))) __bf16 v16bf;
typedef __attribute__((ext_vector_type(8)))  __bf16 v8bf;
typedef __attribute__((ext_vector_type(8)))  float  v8f;

union V16U { v16bf v; v8bf h[2]; };

#define B_DIM 16384
#define E_DIM 1024
#define D_DIM 256
#define H_DIM 16
#define N1_DIM 4096              // H*D columns of GEMM1
#define LN_EPS 1e-5f

// LDS A-tile row stride: 128 + 8 bf16 = 272B = 68 dwords.
// Row r's bank window shifts by 4 dwords -> 16 rows cover all 64 banks
// exactly once per half-wave => conflict-free ds_load_b128 A fragments.
#define LDS_STRIDE 136

// A-matrix fragment (16x32 bf16), contiguous K order (matches load_bfrag):
// lanes 0-15 : row M=lane,    K [kk*32+0 .. +15]
// lanes 16-31: row M=lane-16, K [kk*32+16 .. +31]
__device__ __forceinline__ v16bf load_afrag(const __bf16* rowptr, int kk, int lane) {
    const int base = kk * 32 + ((lane >> 4) << 4);   // +0 or +16
    V16U u;
    u.h[0] = *(const v8bf*)(rowptr + base);
    u.h[1] = *(const v8bf*)(rowptr + base + 8);
    return u.v;
}

// B-matrix fragment (32x16 bf16): lane n<16 holds K=0..15 of column n,
// lane n>=16 holds K=16..31. colptr points at the column's K-row (kbase added).
__device__ __forceinline__ v16bf load_bfrag(const __bf16* colptr, int lane) {
    return *(const v16bf*)(colptr + ((lane >> 4) << 4));
}

__device__ __forceinline__ float wave_red_sum(float v) {
    v += __shfl_xor(v, 1);
    v += __shfl_xor(v, 2);
    v += __shfl_xor(v, 4);
    v += __shfl_xor(v, 8);
    v += __shfl_xor(v, 16);
    return v;
}
__device__ __forceinline__ float half_red_sum(float v) {
    v += __shfl_xor(v, 1);
    v += __shfl_xor(v, 2);
    v += __shfl_xor(v, 4);
    v += __shfl_xor(v, 8);
    return v;
}
__device__ __forceinline__ float half_red_max(float v) {
    v = fmaxf(v, __shfl_xor(v, 1));
    v = fmaxf(v, __shfl_xor(v, 2));
    v = fmaxf(v, __shfl_xor(v, 4));
    v = fmaxf(v, __shfl_xor(v, 8));
    return v;
}

// ---------------------------------------------------------------------------
// Conversion kernels (run once per launch)
// ---------------------------------------------------------------------------
__global__ void conv_x_kernel(const float* __restrict__ x, __bf16* __restrict__ xb) {
    int o = blockIdx.x * blockDim.x + threadIdx.x;
    if (o >= B_DIM * E_DIM) return;
    xb[o] = (__bf16)x[o];
}

// W[h][e][d] f32 -> Wt[(h*256+d)][e] bf16  (column-major for B fragments)
__global__ void conv_W_kernel(const float* __restrict__ W, __bf16* __restrict__ Wt) {
    int o = blockIdx.x * blockDim.x + threadIdx.x;      // index over (h, d, e)
    if (o >= H_DIM * D_DIM * E_DIM) return;
    int h = o / (D_DIM * E_DIM);
    int rem = o - h * (D_DIM * E_DIM);
    int d = rem / E_DIM;
    int e = rem - d * E_DIM;
    Wt[o] = (__bf16)W[(h * E_DIM + e) * D_DIM + d];
}

__global__ void conv_Wout_kernel(const float* __restrict__ Wo, __bf16* __restrict__ Wob) {
    int o = blockIdx.x * blockDim.x + threadIdx.x;
    if (o >= E_DIM * D_DIM) return;
    Wob[o] = (__bf16)Wo[o];
}

// ---------------------------------------------------------------------------
// Kernel 1: GEMM1  h_raw = xb @ Wt^T   (M=16384, N=4096, K=1024)
// WG = 256 threads (8 waves), tile 128(M) x 128(N); K staged in LDS in chunks
// of 128. Each wave: 8 M-tiles x 1 N-tile (acc = 64 VGPRs); one B fragment
// feeds 8 WMMAs (8x reuse -> low L2 pressure, easy latency hiding).
// ---------------------------------------------------------------------------
__global__ void
__launch_bounds__(256)
gemm1_kernel(const __bf16* __restrict__ xb,
             const __bf16* __restrict__ Wt,
             __bf16* __restrict__ h_raw) {
    __shared__ __bf16 As[128 * LDS_STRIDE];    // ~34 KB A chunk [row][k], padded

    const int tid  = threadIdx.x;
    const int wave = tid >> 5;
    const int lane = tid & 31;
    const int half = lane >> 4;
    const int ln16 = lane & 15;

    const int mtile = blockIdx.x >> 5;          // 0..127
    const int ntile = blockIdx.x & 31;          // 0..31
    const int m0 = mtile * 128;
    const int n0 = ntile * 128 + wave * 16;     // this wave's 16 columns

    v8f acc[8];
#pragma unroll
    for (int mt = 0; mt < 8; ++mt) acc[mt] = (v8f){0.f,0.f,0.f,0.f,0.f,0.f,0.f,0.f};

    const __bf16* bcol = Wt + (size_t)(n0 + ln16) * E_DIM;   // column base

    for (int kc = 0; kc < 8; ++kc) {            // K = 1024 in chunks of 128
        __syncthreads();
        // stage A chunk: 128 rows x 128 k bf16; thread copies 128B
        {
            const int row = tid >> 1;
            const int c0  = (tid & 1) * 64;
            const uint4* src = (const uint4*)(xb + (size_t)(m0 + row) * E_DIM + kc * 128 + c0);
            uint4* dst = (uint4*)(&As[row * LDS_STRIDE + c0]);
#pragma unroll
            for (int i = 0; i < 8; ++i) dst[i] = src[i];
        }
        __syncthreads();

#pragma unroll
        for (int kk = 0; kk < 4; ++kk) {
            v16bf bfrag = load_bfrag(bcol + kc * 128 + kk * 32, lane);
#pragma unroll
            for (int mt = 0; mt < 8; ++mt) {
                v16bf afrag = load_afrag(&As[(mt * 16 + ln16) * LDS_STRIDE], kk, lane);
                acc[mt] = __builtin_amdgcn_wmma_f32_16x16x32_bf16(
                    false, afrag, false, bfrag, (short)0, acc[mt], false, false);
            }
        }
    }

    // epilogue: raw h as bf16 (bias/LN/GELU applied in the attention kernel)
#pragma unroll
    for (int mt = 0; mt < 8; ++mt) {
#pragma unroll
        for (int r = 0; r < 8; ++r) {
            const int m = m0 + mt * 16 + r + half * 8;
            h_raw[(size_t)m * N1_DIM + n0 + ln16] = (__bf16)acc[mt][r];
        }
    }
}

// ---------------------------------------------------------------------------
// Kernel 2: per-16-row attention. 512 threads (16 waves = 16 rows).
//   Stage h_raw tile -> LDS, bias+LN+GELU in place, WMMA head-scores,
//   softmax, fold pool_w, emit c2[b,d] (f32).
// ---------------------------------------------------------------------------
__global__ void
__launch_bounds__(512)
attn_kernel(const __bf16* __restrict__ h_raw,
            const float* __restrict__ bias,
            const float* __restrict__ gamma,
            const float* __restrict__ beta,
            const float* __restrict__ pool_w,
            float* __restrict__ c2) {
    __shared__ __bf16 hls[16 * N1_DIM];        // 128 KB: [row][head*256+d]

    const int tid  = threadIdx.x;
    const int wave = tid >> 5;                 // row within tile
    const int lane = tid & 31;
    const int m0   = blockIdx.x * 16;
    const int half = lane >> 4;
    const int ln16 = lane & 15;

    // stage 16 x 4096 bf16 tile (each thread copies 256B)
    {
        const int row = wave;
        const uint4* src = (const uint4*)(h_raw + (size_t)(m0 + row) * N1_DIM + lane * 128);
        uint4* dst = (uint4*)(&hls[row * N1_DIM + lane * 128]);
#pragma unroll
        for (int i = 0; i < 16; ++i) dst[i] = src[i];
    }
    __syncthreads();

    const int m = wave;
    __bf16* hrow = &hls[m * N1_DIM];

    // bias + LayerNorm + GELU per head (8 d-values per lane)
    const int d0 = lane * 8;
#pragma unroll
    for (int h = 0; h < H_DIM; ++h) {
        __bf16* hp = hrow + h * D_DIM;
        float v[8];
        float s = 0.f, s2 = 0.f;
#pragma unroll
        for (int j = 0; j < 8; ++j) {
            v[j] = (float)hp[d0 + j] + bias[h * D_DIM + d0 + j];
            s += v[j]; s2 += v[j] * v[j];
        }
        s  = wave_red_sum(s);
        s2 = wave_red_sum(s2);
        const float mu = s * (1.0f / 256.0f);
        const float var = s2 * (1.0f / 256.0f) - mu * mu;
        const float rstd = rsqrtf(var + LN_EPS);
#pragma unroll
        for (int j = 0; j < 8; ++j) {
            float t = (v[j] - mu) * rstd * gamma[h * D_DIM + d0 + j] + beta[h * D_DIM + d0 + j];
            t = 0.5f * t * (1.0f + erff(t * 0.70710678118654752f));   // exact GELU
            hp[d0 + j] = (__bf16)t;
        }
    }
    __syncthreads();

    // head-scores: 16x16 = h[m] @ h[m]^T via 8 chained WMMAs from LDS
    v8f sc = (v8f){0.f,0.f,0.f,0.f,0.f,0.f,0.f,0.f};
#pragma unroll
    for (int kk = 0; kk < 8; ++kk) {
        v16bf afrag = load_afrag(hrow + (size_t)ln16 * D_DIM, kk, lane);          // row = head
        v16bf bfrag = load_bfrag(hrow + (size_t)ln16 * D_DIM + kk * 32, lane);    // col = head
        sc = __builtin_amdgcn_wmma_f32_16x16x32_bf16(
            false, afrag, false, bfrag, (short)0, sc, false, false);
    }

    // softmax per head-row; fold pool_w: a[g] = sum_h pw[h]*attn[h][g]
    float a_part = 0.f;
#pragma unroll
    for (int r = 0; r < 8; ++r) {
        const float v  = sc[r] * 0.0625f;            // 1/sqrt(256)
        const float mx = half_red_max(v);
        const float e  = expf(v - mx);
        const float sm = half_red_sum(e);
        const float attn = e / sm;
        a_part += pool_w[r + half * 8] * attn;
    }
    const float a_full = a_part + __shfl_xor(a_part, 16);

    // c2[m][d] = sum_g a[g] * h[m][g][d]
    float cacc[8] = {0.f,0.f,0.f,0.f,0.f,0.f,0.f,0.f};
#pragma unroll
    for (int g = 0; g < 16; ++g) {
        const float ag = __shfl(a_full, g);
        const __bf16* hv = hrow + g * D_DIM + d0;
#pragma unroll
        for (int j = 0; j < 8; ++j) cacc[j] += ag * (float)hv[j];
    }
    float* c2p = c2 + (size_t)(m0 + m) * D_DIM + d0;
#pragma unroll
    for (int j = 0; j < 8; ++j) c2p[j] = cacc[j];
}

// ---------------------------------------------------------------------------
// Kernel 3: out = x + c2 @ Wout^T  (M=16384, N=1024, K=256)
// Same 128x128 tiling as gemm1; A (c2, f32) converted to bf16 during staging.
// ---------------------------------------------------------------------------
__global__ void
__launch_bounds__(256)
proj_add_kernel(const float* __restrict__ x,
                const float* __restrict__ c2,
                const __bf16* __restrict__ Wob,
                float* __restrict__ out) {
    __shared__ __bf16 As[128 * LDS_STRIDE];    // padded

    const int tid  = threadIdx.x;
    const int wave = tid >> 5;
    const int lane = tid & 31;
    const int half = lane >> 4;
    const int ln16 = lane & 15;

    const int mtile = blockIdx.x >> 3;          // 0..127
    const int ntile = blockIdx.x & 7;           // 0..7
    const int m0 = mtile * 128;
    const int n0 = ntile * 128 + wave * 16;

    v8f acc[8];
#pragma unroll
    for (int mt = 0; mt < 8; ++mt) acc[mt] = (v8f){0.f,0.f,0.f,0.f,0.f,0.f,0.f,0.f};

    const __bf16* bcol = Wob + (size_t)(n0 + ln16) * D_DIM;

    for (int kc = 0; kc < 2; ++kc) {            // K = 256 in chunks of 128
        __syncthreads();
        // stage A chunk from f32 c2, converting to bf16 (vectorized)
        {
            const int row = tid >> 1;
            const int c0  = (tid & 1) * 64;
            const float4* src = (const float4*)(c2 + (size_t)(m0 + row) * D_DIM + kc * 128 + c0);
            uint4* dst = (uint4*)(&As[row * LDS_STRIDE + c0]);
#pragma unroll
            for (int i = 0; i < 8; ++i) {
                float4 f0 = src[2 * i];
                float4 f1 = src[2 * i + 1];
                union { __bf16 b[8]; uint4 u; } pk;
                pk.b[0] = (__bf16)f0.x; pk.b[1] = (__bf16)f0.y;
                pk.b[2] = (__bf16)f0.z; pk.b[3] = (__bf16)f0.w;
                pk.b[4] = (__bf16)f1.x; pk.b[5] = (__bf16)f1.y;
                pk.b[6] = (__bf16)f1.z; pk.b[7] = (__bf16)f1.w;
                dst[i] = pk.u;
            }
        }
        __syncthreads();

#pragma unroll
        for (int kk = 0; kk < 4; ++kk) {
            v16bf bfrag = load_bfrag(bcol + kc * 128 + kk * 32, lane);
#pragma unroll
            for (int mt = 0; mt < 8; ++mt) {
                v16bf afrag = load_afrag(&As[(mt * 16 + ln16) * LDS_STRIDE], kk, lane);
                acc[mt] = __builtin_amdgcn_wmma_f32_16x16x32_bf16(
                    false, afrag, false, bfrag, (short)0, acc[mt], false, false);
            }
        }
    }

#pragma unroll
    for (int mt = 0; mt < 8; ++mt) {
#pragma unroll
        for (int r = 0; r < 8; ++r) {
            const int m = m0 + mt * 16 + r + half * 8;
            const size_t idx = (size_t)m * E_DIM + n0 + ln16;
            out[idx] = x[idx] + acc[mt][r];
        }
    }
}

// ---------------------------------------------------------------------------
// Launch
// ---------------------------------------------------------------------------
extern "C" void kernel_launch(void* const* d_in, const int* in_sizes, int n_in,
                              void* d_out, int out_size, void* d_ws, size_t ws_size,
                              hipStream_t stream) {
    const float* x      = (const float*)d_in[0];
    const float* W      = (const float*)d_in[1];
    const float* b      = (const float*)d_in[2];
    const float* gamma  = (const float*)d_in[3];
    const float* beta   = (const float*)d_in[4];
    const float* Wout   = (const float*)d_in[5];
    const float* pool_w = (const float*)d_in[6];
    float* out          = (float*)d_out;

    char* ws = (char*)d_ws;
    __bf16* xb    = (__bf16*)(ws);                                     // 32 MB
    __bf16* Wt    = (__bf16*)(ws + (size_t)33554432);                  // 8 MB
    __bf16* Wob   = (__bf16*)(ws + (size_t)33554432 + 8388608);        // 0.5 MB
    __bf16* h_raw = (__bf16*)(ws + (size_t)33554432 + 8388608 + 524288);        // 128 MB
    float*  c2    = (float*)(ws + (size_t)33554432 + 8388608 + 524288 + 134217728); // 16 MB

    // 0) one-time conversions
    conv_x_kernel<<<(B_DIM * E_DIM) / 256, 256, 0, stream>>>(x, xb);
    conv_W_kernel<<<(H_DIM * D_DIM * E_DIM) / 256, 256, 0, stream>>>(W, Wt);
    conv_Wout_kernel<<<(E_DIM * D_DIM) / 256, 256, 0, stream>>>(Wout, Wob);

    // 1) h_raw = xb @ Wt^T        (grid: 128 Mtiles x 32 Ntiles)
    gemm1_kernel<<<128 * 32, 256, 0, stream>>>(xb, Wt, h_raw);

    // 2) bias+LN+GELU + head attention + pool-fold -> c2
    attn_kernel<<<B_DIM / 16, 512, 0, stream>>>(h_raw, b, gamma, beta, pool_w, c2);

    // 3) out = x + c2 @ Wout^T    (grid: 128 Mtiles x 8 Ntiles)
    proj_add_kernel<<<128 * 8, 256, 0, stream>>>(x, c2, Wob, out);
}
